// Block_730144440514
// MI455X (gfx1250) — compile-verified, matrix operand order checked by reading, not run on hardware
//
#include <hip/hip_runtime.h>
#include <hip/hip_bf16.h>

// ---------------- problem constants (from reference) ----------------
static constexpr int   kH = 1024;     // hidden
static constexpr int   kN = 8;        // experts
static constexpr int   kK = 2;        // top-k
static constexpr int   kR = 8;        // lora rank
static constexpr int   kV = 32000;    // vocab
static constexpr int   kI = 128;      // expert intermediate
static constexpr int   kT = 4096;     // B*S tokens
static constexpr float kScale = 16.0f / 8.0f;  // alpha/r = 2.0
static constexpr float kEps = 1.1920928955078125e-07f;

typedef __attribute__((ext_vector_type(16))) _Float16 v16h;
typedef __attribute__((ext_vector_type(8)))  float    v8f;
typedef __attribute__((ext_vector_type(4)))  _Float16 h4;

union Frag { v16h v; uint4 q[2]; };

// ---------------- block-wide reduction (wave32) ----------------
__device__ __forceinline__ float blockReduceBcast(float v, volatile float* red) {
    const int tid  = threadIdx.x;
    const int lane = tid & 31;
    const int wid  = tid >> 5;
    #pragma unroll
    for (int o = 16; o > 0; o >>= 1) v += __shfl_down(v, o, 32);
    if (lane == 0) red[wid] = v;
    __syncthreads();
    if (tid == 0) {
        float s = 0.f;
        #pragma unroll
        for (int i = 0; i < 8; ++i) s += red[i];
        red[8] = s;
    }
    __syncthreads();
    float r = red[8];
    __syncthreads();   // safe reuse of red[] by next reduction
    return r;
}

// ---------------- kernel 1: LoRA chain + residual + RMSNorm1 + router top2 ----------------
__global__ void k_lora_rms_router(
    const float* __restrict__ x,
    const float* __restrict__ qA, const float* __restrict__ qB,
    const float* __restrict__ kA, const float* __restrict__ kB,
    const float* __restrict__ vA, const float* __restrict__ vB,
    const float* __restrict__ oA, const float* __restrict__ oB,
    const float* __restrict__ n1w, const float* __restrict__ rw,
    float* __restrict__ x1f, _Float16* __restrict__ x1h,
    int* __restrict__ topIdx, float* __restrict__ topW)
{
    __shared__ float red[9];
    __shared__ float logits[kN];
    const int t   = blockIdx.x;
    const int tid = threadIdx.x;
    const float* xr = x + (size_t)t * kH;

    float xv[4]; int hh[4];
    #pragma unroll
    for (int j = 0; j < 4; ++j) { hh[j] = tid + j * 256; xv[j] = xr[hh[j]]; }

    // x @ {q,k,v}_A^T   (rank-8 projections, block reductions)
    float tq[kR], tk[kR], tv[kR];
    #pragma unroll
    for (int r = 0; r < kR; ++r) {
        float aq = 0.f, ak = 0.f, av = 0.f;
        #pragma unroll
        for (int j = 0; j < 4; ++j) {
            aq += xv[j] * qA[r * kH + hh[j]];
            ak += xv[j] * kA[r * kH + hh[j]];
            av += xv[j] * vA[r * kH + hh[j]];
        }
        tq[r] = blockReduceBcast(aq, red);
        tk[r] = blockReduceBcast(ak, red);
        tv[r] = blockReduceBcast(av, red);
    }

    // s = scale*(tq@qB^T + tk@kB^T + tv@vB^T)
    float sv[4];
    #pragma unroll
    for (int j = 0; j < 4; ++j) {
        float a = 0.f;
        #pragma unroll
        for (int r = 0; r < kR; ++r) {
            a += tq[r] * qB[hh[j] * kR + r]
               + tk[r] * kB[hh[j] * kR + r]
               + tv[r] * vB[hh[j] * kR + r];
        }
        sv[j] = kScale * a;
    }

    // o-LoRA: to = s @ oA^T ; a = scale * (to @ oB^T)
    float to[kR];
    #pragma unroll
    for (int r = 0; r < kR; ++r) {
        float a = 0.f;
        #pragma unroll
        for (int j = 0; j < 4; ++j) a += sv[j] * oA[r * kH + hh[j]];
        to[r] = blockReduceBcast(a, red);
    }

    float yv[4]; float vp = 0.f;
    #pragma unroll
    for (int j = 0; j < 4; ++j) {
        float a = 0.f;
        #pragma unroll
        for (int r = 0; r < kR; ++r) a += to[r] * oB[hh[j] * kR + r];
        yv[j] = xv[j] + kScale * a;
        vp += yv[j] * yv[j];
    }

    const float var = blockReduceBcast(vp, red) * (1.0f / kH);
    const float rs  = rsqrtf(var + kEps);

    float x1v[4];
    #pragma unroll
    for (int j = 0; j < 4; ++j) {
        x1v[j] = yv[j] * rs * n1w[hh[j]];
        x1f[(size_t)t * kH + hh[j]] = x1v[j];
        x1h[(size_t)t * kH + hh[j]] = (_Float16)x1v[j];
    }

    // router logits + top-2 softmax
    #pragma unroll
    for (int e = 0; e < kN; ++e) {
        float a = 0.f;
        #pragma unroll
        for (int j = 0; j < 4; ++j) a += x1v[j] * rw[e * kH + hh[j]];
        float le = blockReduceBcast(a, red);
        if (tid == 0) logits[e] = le;
    }
    __syncthreads();
    if (tid == 0) {
        int i0 = 0;
        for (int e = 1; e < kN; ++e) if (logits[e] > logits[i0]) i0 = e;
        int i1 = -1;
        for (int e = 0; e < kN; ++e)
            if (e != i0 && (i1 < 0 || logits[e] > logits[i1])) i1 = e;
        const float m  = logits[i0];
        const float e0 = 1.0f;
        const float e1 = expf(logits[i1] - m);
        const float inv = 1.0f / (e0 + e1);
        topIdx[2 * t + 0] = i0;  topIdx[2 * t + 1] = i1;
        topW [2 * t + 0] = e0 * inv;  topW[2 * t + 1] = e1 * inv;
    }
}

// ---------------- f32 -> f16 convert ----------------
__global__ void k_f32_to_f16(const float* __restrict__ src, _Float16* __restrict__ dst, long n4) {
    long i = (long)blockIdx.x * blockDim.x + threadIdx.x;
    if (i >= n4) return;
    const float4 v = ((const float4*)src)[i];
    h4 o; o[0] = (_Float16)v.x; o[1] = (_Float16)v.y; o[2] = (_Float16)v.z; o[3] = (_Float16)v.w;
    ((h4*)dst)[i] = o;
}

// ---------------- CDNA5 async global->LDS staging ----------------
// LDS byte offset of a shared pointer: flat LDS aperture has zero low 32 bits,
// hardware maps flat->LDS by truncating to addr[31:0] (ISA 10.2), so the low
// 32 bits of the generic address ARE the wave-relative LDS offset.
__device__ __forceinline__ unsigned ldsOffset(const void* p) {
    return (unsigned)(uintptr_t)p;
}

__device__ __forceinline__ void asyncCopy16(unsigned ldsOff, const void* gptr) {
    unsigned long long ga = (unsigned long long)(uintptr_t)gptr;
    asm volatile("global_load_async_to_lds_b128 %0, %1, off"
                 :: "v"(ldsOff), "v"(ga) : "memory");
}

__device__ __forceinline__ void waitAsync0() {
    asm volatile("s_wait_asynccnt 0x0" ::: "memory");
}

// ---------------- 64x128 WMMA GEMM tile: C = A[64,K] * B[128,K]^T ----------------
// 256 threads = 8 waves in a 2(M) x 4(N) grid; each wave owns 32x32 = 2x2 16x16 frags.
// K-step 64, double-buffered LDS, async DMA staging (ASYNCcnt) overlapped with WMMA.
// LDS rows padded to 144B (64 f16 = 128B data) so every fragment half is one 16B ds_read.
static constexpr int kRowB   = 144;                 // padded row stride (bytes)
static constexpr int kABytes = 64  * kRowB;         // 9216
static constexpr int kBBytes = 128 * kRowB;         // 18432
static constexpr int kBufB   = kABytes + kBBytes;   // 27648 per buffer
static constexpr int kGemmLds = 2 * kBufB;          // 55296 B (of 320 KB/WGP)

__device__ __forceinline__ void issueTile(
    const _Float16* __restrict__ A, const _Float16* __restrict__ B,
    int ldA, int ldB, int k0, unsigned ldsBase, int tid)
{
    // A tile: 64 rows x 64 f16 = 8 KB -> 2 x b128 per thread
    #pragma unroll
    for (int it = 0; it < 2; ++it) {
        const int idx = tid + it * 256;
        const int row = idx >> 3, cg = idx & 7;
        asyncCopy16(ldsBase + row * kRowB + cg * 16,
                    A + (size_t)row * ldA + k0 + cg * 8);
    }
    // B tile: 128 rows x 64 f16 = 16 KB -> 4 x b128 per thread
    #pragma unroll
    for (int it = 0; it < 4; ++it) {
        const int idx = tid + it * 256;
        const int row = idx >> 3, cg = idx & 7;
        asyncCopy16(ldsBase + kABytes + row * kRowB + cg * 16,
                    B + (size_t)row * ldB + k0 + cg * 8);
    }
}

__device__ __forceinline__ void gemm_64x128_f16(
    const _Float16* __restrict__ A, const _Float16* __restrict__ B,
    int K, int ldA, int ldB, char* smem, v8f acc[2][2])
{
    const int tid  = threadIdx.x;
    const int lane = tid & 31;
    const int wid  = tid >> 5;
    const int wm   = wid >> 2;        // 0..1
    const int wn   = wid & 3;         // 0..3
    const int half = lane >> 4;       // 0..1
    const int l16  = lane & 15;

    #pragma unroll
    for (int mt = 0; mt < 2; ++mt)
        #pragma unroll
        for (int nt = 0; nt < 2; ++nt)
            #pragma unroll
            for (int r = 0; r < 8; ++r) acc[mt][nt][r] = 0.f;

    const unsigned ldsBase = ldsOffset(smem);
    const int nIter = K >> 6;                 // K-step 64

    issueTile(A, B, ldA, ldB, 0, ldsBase, tid);

    for (int i = 0; i < nIter; ++i) {
        const char* buf = smem + (i & 1) * kBufB;

        waitAsync0();          // this wave's DMA into current buffer done
        __syncthreads();       // every wave's DMA visible

        if (i + 1 < nIter)     // prefetch next K-step into the other buffer
            issueTile(A, B, ldA, ldB, (i + 1) << 6,
                      ldsBase + ((i + 1) & 1) * kBufB, tid);

        #pragma unroll
        for (int s = 0; s < 2; ++s) {          // two K=32 sub-steps
            Frag aF[2], bF[2];
            #pragma unroll
            for (int mt = 0; mt < 2; ++mt) {
                const int m = wm * 32 + mt * 16 + l16;
                const char* p = buf + m * kRowB + s * 64 + half * 16;
                aF[mt].q[0] = *(const uint4*)p;          // K = s*32 + half*8 ..
                aF[mt].q[1] = *(const uint4*)(p + 32);   // K = s*32 + 16 + half*8 ..
            }
            #pragma unroll
            for (int nt = 0; nt < 2; ++nt) {
                const int n = wn * 32 + nt * 16 + l16;
                const char* p = buf + kABytes + n * kRowB + s * 64 + half * 16;
                bF[nt].q[0] = *(const uint4*)p;
                bF[nt].q[1] = *(const uint4*)(p + 32);
            }
            #pragma unroll
            for (int mt = 0; mt < 2; ++mt)
                #pragma unroll
                for (int nt = 0; nt < 2; ++nt)
                    acc[mt][nt] = __builtin_amdgcn_wmma_f32_16x16x32_f16(
                        false, aF[mt].v, false, bF[nt].v,
                        (short)0, acc[mt][nt], false, false);
        }
        __syncthreads();       // all waves done reading buf before it is re-filled
    }
}

// ---------------- MoE gate GEMM: g[e,t,i] = silu(x1 @ eg_w[e]^T + eg_b[e]) ----------------
__global__ void k_gate(const _Float16* __restrict__ x1h, const _Float16* __restrict__ egwh,
                       const float* __restrict__ egb, _Float16* __restrict__ g)
{
    extern __shared__ char smem[];
    const int e  = blockIdx.y;
    const int bM = blockIdx.x * 64;
    v8f acc[2][2];
    gemm_64x128_f16(x1h + (size_t)bM * kH, egwh + (size_t)e * kI * kH,
                    kH, kH, kH, smem, acc);

    const int tid = threadIdx.x, lane = tid & 31, wid = tid >> 5;
    const int wm = wid >> 2, wn = wid & 3, half = lane >> 4, l16 = lane & 15;
    #pragma unroll
    for (int mt = 0; mt < 2; ++mt)
        #pragma unroll
        for (int nt = 0; nt < 2; ++nt)
            #pragma unroll
            for (int r = 0; r < 8; ++r) {
                const int row = bM + wm * 32 + mt * 16 + r + 8 * half;   // token
                const int col = wn * 32 + nt * 16 + l16;                 // 0..127 (I)
                float v = acc[mt][nt][r] + egb[e * kI + col];
                float s = v / (1.0f + expf(-v));                          // silu
                g[((size_t)e * kT + row) * kI + col] = (_Float16)s;
            }
}

// ---------------- MoE combine (top-2 up-proj) + residual + RMSNorm2 ----------------
__global__ void k_combine(const float* __restrict__ x1f, const _Float16* __restrict__ g,
                          const float* __restrict__ euw, const float* __restrict__ eub,
                          const int* __restrict__ topIdx, const float* __restrict__ topW,
                          const float* __restrict__ n2w, _Float16* __restrict__ x2h)
{
    __shared__ float red[9];
    __shared__ float g0s[kI], g1s[kI];
    __shared__ int   sidx[2];
    __shared__ float sw[2];
    const int t = blockIdx.x, tid = threadIdx.x;
    if (tid < 2) { sidx[tid] = topIdx[2 * t + tid]; sw[tid] = topW[2 * t + tid]; }
    __syncthreads();
    const int e0 = sidx[0], e1 = sidx[1];
    const float w0 = sw[0], w1 = sw[1];
    if (tid < kI) {
        g0s[tid] = (float)g[((size_t)e0 * kT + t) * kI + tid];
        g1s[tid] = (float)g[((size_t)e1 * kT + t) * kI + tid];
    }
    __syncthreads();

    float zv[4]; float vp = 0.f;
    #pragma unroll
    for (int j = 0; j < 4; ++j) {
        const int h = tid + j * 256;
        const float* w0p = euw + ((size_t)e0 * kH + h) * kI;
        const float* w1p = euw + ((size_t)e1 * kH + h) * kI;
        __builtin_prefetch(w0p, 0, 1);   // global_prefetch_b8
        __builtin_prefetch(w1p, 0, 1);
        float a0 = 0.f, a1 = 0.f;
        for (int i = 0; i < kI; ++i) { a0 += g0s[i] * w0p[i]; a1 += g1s[i] * w1p[i]; }
        const float y = w0 * (a0 + eub[e0 * kH + h]) + w1 * (a1 + eub[e1 * kH + h]);
        zv[j] = x1f[(size_t)t * kH + h] + y;
        vp += zv[j] * zv[j];
    }
    const float var = blockReduceBcast(vp, red) * (1.0f / kH);
    const float rs  = rsqrtf(var + kEps);
    #pragma unroll
    for (int j = 0; j < 4; ++j) {
        const int h = tid + j * 256;
        x2h[(size_t)t * kH + h] = (_Float16)(zv[j] * rs * n2w[h]);
    }
}

// ---------------- LM head: out[T,V] = x2 @ head_w^T ----------------
__global__ void k_head(const _Float16* __restrict__ x2h, const _Float16* __restrict__ wh,
                       float* __restrict__ out)
{
    extern __shared__ char smem[];
    const int bM = blockIdx.y * 64;
    const int bN = blockIdx.x * 128;
    v8f acc[2][2];
    gemm_64x128_f16(x2h + (size_t)bM * kH, wh + (size_t)bN * kH,
                    kH, kH, kH, smem, acc);

    const int tid = threadIdx.x, lane = tid & 31, wid = tid >> 5;
    const int wm = wid >> 2, wn = wid & 3, half = lane >> 4, l16 = lane & 15;
    #pragma unroll
    for (int mt = 0; mt < 2; ++mt)
        #pragma unroll
        for (int nt = 0; nt < 2; ++nt)
            #pragma unroll
            for (int r = 0; r < 8; ++r) {
                const int row = bM + wm * 32 + mt * 16 + r + 8 * half;
                const int col = bN + wn * 32 + nt * 16 + l16;
                out[(size_t)row * kV + col] = acc[mt][nt][r];
            }
}

// ---------------- host launcher ----------------
extern "C" void kernel_launch(void* const* d_in, const int* in_sizes, int n_in,
                              void* d_out, int out_size, void* d_ws, size_t ws_size,
                              hipStream_t stream)
{
    const float* x    = (const float*)d_in[0];
    const float* qA   = (const float*)d_in[1];
    const float* qB   = (const float*)d_in[2];
    const float* kA   = (const float*)d_in[3];
    const float* kB   = (const float*)d_in[4];
    const float* vA   = (const float*)d_in[5];
    const float* vB   = (const float*)d_in[6];
    const float* oA   = (const float*)d_in[7];
    const float* oB   = (const float*)d_in[8];
    const float* n1w  = (const float*)d_in[9];
    const float* n2w  = (const float*)d_in[10];
    const float* rw   = (const float*)d_in[11];
    const float* egw  = (const float*)d_in[12];
    const float* egb  = (const float*)d_in[13];
    const float* euw  = (const float*)d_in[14];
    const float* eub  = (const float*)d_in[15];
    const float* hw   = (const float*)d_in[16];

    char* ws = (char*)d_ws;
    size_t off = 0;
    auto alloc = [&](size_t bytes) -> void* {
        void* p = ws + off;
        off = (off + bytes + 255) & ~(size_t)255;
        return p;
    };
    float*    x1f    = (float*)   alloc((size_t)kT * kH * 4);
    _Float16* x1h    = (_Float16*)alloc((size_t)kT * kH * 2);
    _Float16* x2h    = (_Float16*)alloc((size_t)kT * kH * 2);
    _Float16* hwh    = (_Float16*)alloc((size_t)kV * kH * 2);
    _Float16* egwh   = (_Float16*)alloc((size_t)kN * kI * kH * 2);
    _Float16* g      = (_Float16*)alloc((size_t)kN * kT * kI * 2);
    int*      topIdx = (int*)     alloc((size_t)kT * 2 * 4);
    float*    topW   = (float*)   alloc((size_t)kT * 2 * 4);
    (void)ws_size; (void)in_sizes; (void)n_in; (void)out_size;

    // 1) fused LoRA + RMSNorm1 + router
    k_lora_rms_router<<<kT, 256, 0, stream>>>(x, qA, qB, kA, kB, vA, vB, oA, oB,
                                              n1w, rw, x1f, x1h, topIdx, topW);

    // 2) one-time weight converts to f16
    {
        long n4 = (long)kV * kH / 4;
        k_f32_to_f16<<<(int)((n4 + 255) / 256), 256, 0, stream>>>(hw, hwh, n4);
    }
    {
        long n4 = (long)kN * kI * kH / 4;
        k_f32_to_f16<<<(int)((n4 + 255) / 256), 256, 0, stream>>>(egw, egwh, n4);
    }

    // 3) dense gate GEMM (WMMA + async LDS staging), all experts
    {
        dim3 grid(kT / 64, kN);
        k_gate<<<grid, 256, kGemmLds, stream>>>(x1h, egwh, egb, g);
    }

    // 4) top-2 up-projection + residual + RMSNorm2
    k_combine<<<kT, 256, 0, stream>>>(x1f, g, euw, eub, topIdx, topW, n2w, x2h);

    // 5) LM head GEMM (WMMA + async LDS staging) -> fp32 logits
    {
        dim3 grid(kV / 128, kT / 64);
        k_head<<<grid, 256, kGemmLds, stream>>>(x2h, hwh, (float*)d_out);
    }
}
